// my_model6_74998718923527
// MI455X (gfx1250) — compile-verified, matrix operand order checked by reading, not run on hardware
//
#include <hip/hip_runtime.h>
#include <hip/hip_bf16.h>

typedef __attribute__((ext_vector_type(16))) _Float16 v16h;
typedef __attribute__((ext_vector_type(8)))  _Float16 v8h;
typedef __attribute__((ext_vector_type(4)))  _Float16 h4;
typedef __attribute__((ext_vector_type(8)))  float    v8f;

constexpr int Bn  = 4;
constexpr int Ln  = 1024;
constexpr int Dn  = 768;
constexpr int Hn  = 12;
constexpr int BSn = 64;
constexpr int Mn  = 48;
constexpr int NEn = 24;
constexpr int Pn  = NEn * (NEn - 1);   // 552
constexpr int En  = Pn + NEn;          // 576 edges incl self loops
constexpr int Rn  = 97;
constexpr int Nrow = Bn * Pn;          // 2208

// WMMA f16 A/B fragment permutation: logical k in [0,32) -> grp*16+hi slot.
// Inverse of k = (hi<8 ? hi : hi+8) + grp*8. Aligned 4-chunks of k stay contiguous.
__device__ __forceinline__ int perm_idx(int k) {
    return (((k >> 3) & 1) << 4) | ((k & 7) | ((k >> 4) << 3));
}
__device__ __forceinline__ v16h frag_load(const _Float16* p) {
    v8h lo = *(const v8h*)p;
    v8h hi = *(const v8h*)(p + 8);
    return __builtin_shufflevector(lo, hi, 0, 1, 2, 3, 4, 5, 6, 7,
                                   8, 9, 10, 11, 12, 13, 14, 15);
}

// ---------------- stage 1: logsumexp entity pooling ----------------
__global__ __launch_bounds__(256)
void k_entity(const float* __restrict__ context, const int* __restrict__ mmap,
              const float* __restrict__ emap, float* __restrict__ entity,
              float* __restrict__ esum) {
    int b = blockIdx.x / NEn, n = blockIdx.x % NEn;
    if (threadIdx.x == 0) {
        float s = 0.f;
        for (int m = 0; m < Mn; ++m) s += emap[(b * NEn + n) * Mn + m];
        esum[b * NEn + n] = s;
    }
    for (int d = threadIdx.x; d < Dn; d += blockDim.x) {
        float acc = 0.f;
        for (int m = 0; m < Mn; ++m) {
            float w = emap[(b * NEn + n) * Mn + m];
            if (w != 0.f) {
                int l = mmap[b * Mn + m];
                acc += w * __expf(context[((long long)b * Ln + l) * Dn + d]);
            }
        }
        entity[((long long)b * NEn + n) * Dn + d] = __logf(acc);
    }
}

// ---------------- stage 2: cur[b,h,n,l] ----------------
__global__ __launch_bounds__(256)
void k_cur(const float* __restrict__ att, const int* __restrict__ mmap,
           const float* __restrict__ emap, const float* __restrict__ esum,
           float* __restrict__ cur) {
    int idx = blockIdx.x;
    int n = idx % NEn, hh = (idx / NEn) % Hn, b = idx / (NEn * Hn);
    float inv = 1.f / (esum[b * NEn + n] + 1e-20f);
    for (int l = threadIdx.x; l < Ln; l += blockDim.x) {
        float acc = 0.f;
        for (int m = 0; m < Mn; ++m) {
            float w = emap[(b * NEn + n) * Mn + m];
            if (w != 0.f) {
                int ml = mmap[b * Mn + m];
                acc += w * att[(((long long)b * Hn + hh) * Ln + ml) * Ln + l];
            }
        }
        cur[(((long long)b * Hn + hh) * NEn + n) * Ln + l] = acc * inv;
    }
}

// ---------------- stage 3: ca[b,p,l] + row normalize ----------------
__global__ __launch_bounds__(256)
void k_ca(const float* __restrict__ cur, const int* __restrict__ hts,
          float* __restrict__ ca) {
    int b = blockIdx.x / Pn, p = blockIdx.x % Pn;
    int h0 = hts[p], h1 = hts[Pn + p];
    __shared__ float red[256];
    float* cab = ca + (long long)(b * Pn + p) * Ln;
    float lsum = 0.f;
    for (int l = threadIdx.x; l < Ln; l += 256) {
        float s = 0.f;
        for (int hh = 0; hh < Hn; ++hh) {
            const float* cb = cur + ((long long)b * Hn + hh) * NEn * Ln;
            s += cb[h0 * Ln + l] * cb[h1 * Ln + l];
        }
        cab[l] = s;
        lsum += s;
    }
    red[threadIdx.x] = lsum;
    __syncthreads();
    for (int off = 128; off > 0; off >>= 1) {
        if (threadIdx.x < off) red[threadIdx.x] += red[threadIdx.x + off];
        __syncthreads();
    }
    float inv = 1.f / (red[0] + 1e-20f);
    for (int l = threadIdx.x; l < Ln; l += 256) cab[l] *= inv;
}

// ---------------- generic WMMA GEMM: C = act(A @ W + bias) ----------------
// A [M,K] f32 row-major, W [K,N] f32 row-major, f16 compute / f32 accumulate.
// LDS tiles stored in WMMA fragment-permuted layout -> ds_load_b128 operand fetch.
#define GM_BM 64
#define GM_BN 64
#define GM_KT 32
#define GM_LDH 40   // padded row stride (halves); 80B keeps 16B alignment
__global__ __launch_bounds__(256)
void k_gemm(const float* __restrict__ A, const float* __restrict__ W,
            const float* __restrict__ bias, float* __restrict__ C,
            int Mrows, int K, int N,
            long long sA, long long sW, long long sC, int act) {
    __shared__ __align__(16) _Float16 Ap[GM_BM][GM_LDH];
    __shared__ __align__(16) _Float16 Bp[GM_BN][GM_LDH];
    int bz = blockIdx.z;
    A += (long long)bz * sA;  W += (long long)bz * sW;  C += (long long)bz * sC;
    int m_base = blockIdx.y * GM_BM;
    int n_base = blockIdx.x * GM_BN;
    int tid = threadIdx.x, wave = tid >> 5, lane = tid & 31;
    int wm = wave >> 1, wn = wave & 1;
    int grp = lane >> 4, l15 = lane & 15;
    v8f acc0 = {}, acc1 = {};
    for (int k0 = 0; k0 < K; k0 += GM_KT) {
        // ---- stage A tile: 64 rows x 8 float4 chunks -> 2 chunks/thread ----
        #pragma unroll
        for (int it = 0; it < 2; ++it) {
            int i = tid + it * 256;
            int r = i >> 3, cc = (i & 7) * 4;
            int gm = m_base + r;
            float4 v = make_float4(0.f, 0.f, 0.f, 0.f);
            if (gm < Mrows) {
                const float* ap = A + (long long)gm * K + k0 + cc;
                v = *(const float4*)ap;
                if (k0 + GM_KT < K) __builtin_prefetch(ap + GM_KT, 0, 3);
            }
            h4 h = { (_Float16)v.x, (_Float16)v.y, (_Float16)v.z, (_Float16)v.w };
            *(h4*)&Ap[r][perm_idx(cc)] = h;
        }
        // ---- stage B tile: 64 cols x 8 k-chunks (4 strided rows) ----
        #pragma unroll
        for (int it = 0; it < 2; ++it) {
            int i = tid + it * 256;
            int c = i >> 3, kc = (i & 7) * 4;
            int gn = n_base + c;
            float v0 = 0.f, v1 = 0.f, v2 = 0.f, v3 = 0.f;
            if (gn < N) {
                const float* wp = W + (long long)(k0 + kc) * N + gn;
                v0 = wp[0]; v1 = wp[N]; v2 = wp[2 * N]; v3 = wp[3 * N];
                if (k0 + GM_KT < K) __builtin_prefetch(wp + (long long)GM_KT * N, 0, 3);
            }
            h4 h = { (_Float16)v0, (_Float16)v1, (_Float16)v2, (_Float16)v3 };
            *(h4*)&Bp[c][perm_idx(kc)] = h;
        }
        __syncthreads();
        v16h af  = frag_load(&Ap[wm * 16 + l15][grp * 16]);
        v16h bf0 = frag_load(&Bp[wn * 32 + l15][grp * 16]);
        v16h bf1 = frag_load(&Bp[wn * 32 + 16 + l15][grp * 16]);
        acc0 = __builtin_amdgcn_wmma_f32_16x16x32_f16(false, af, false, bf0,
                                                      (short)0, acc0, false, false);
        acc1 = __builtin_amdgcn_wmma_f32_16x16x32_f16(false, af, false, bf1,
                                                      (short)0, acc1, false, false);
        __syncthreads();
    }
    #pragma unroll
    for (int r = 0; r < 8; ++r) {
        int gm = m_base + wm * 16 + grp * 8 + r;           // ISA C/D f32 layout
        if (gm >= Mrows) continue;
        int gn0 = n_base + wn * 32 + l15, gn1 = gn0 + 16;
        if (gn0 < N) {
            float o = acc0[r] + (bias ? bias[gn0] : 0.f);
            C[(long long)gm * N + gn0] = act ? tanhf(o) : o;
        }
        if (gn1 < N) {
            float o = acc1[r] + (bias ? bias[gn1] : 0.f);
            C[(long long)gm * N + gn1] = act ? tanhf(o) : o;
        }
    }
}

// ---------------- fused group-bilinear WMMA ----------------
// out[n,r] = sum_{h,i,j} a1[n,h*64+i]*a2[n,h*64+j]*W[((h*64+i)*64+j), r] + bias[r]
// Outer-product K-tiles generated on the fly, directly in fragment-permuted layout.
#define GB_ROWS 16
#define GB_COLS 64
#define GB_KT   32
__global__ __launch_bounds__(128)
void k_gbl(const float* __restrict__ a1, const float* __restrict__ a2,
           const float* __restrict__ W, const float* __restrict__ bias,
           float* __restrict__ out, int NrowsArg, int R) {
    __shared__ __align__(16) _Float16 A1s[GB_ROWS][Dn];
    __shared__ __align__(16) _Float16 A2s[GB_ROWS][Dn];
    __shared__ __align__(16) _Float16 Xp[GB_ROWS][GM_LDH];
    __shared__ __align__(16) _Float16 Wp[GB_COLS][GM_LDH];
    int row_base = blockIdx.y * GB_ROWS;
    int col_base = blockIdx.x * GB_COLS;
    int tid = threadIdx.x, wave = tid >> 5, lane = tid & 31;
    int grp = lane >> 4, l15 = lane & 15;
    // preload a1/a2 rows as f16 (float4 vector loads, h4 LDS stores)
    for (int i = tid; i < GB_ROWS * (Dn / 4); i += 128) {
        int r = i / (Dn / 4), c4 = (i % (Dn / 4)) * 4;
        int gr = row_base + r;
        float4 u = make_float4(0.f, 0.f, 0.f, 0.f), v = u;
        if (gr < NrowsArg) {
            u = *(const float4*)(a1 + (long long)gr * Dn + c4);
            v = *(const float4*)(a2 + (long long)gr * Dn + c4);
        }
        h4 hu = { (_Float16)u.x, (_Float16)u.y, (_Float16)u.z, (_Float16)u.w };
        h4 hv = { (_Float16)v.x, (_Float16)v.y, (_Float16)v.z, (_Float16)v.w };
        *(h4*)&A1s[r][c4] = hu;
        *(h4*)&A2s[r][c4] = hv;
    }
    __syncthreads();
    v8f acc = {};
    const int K = Hn * BSn * BSn;   // 49152
    for (int k0 = 0; k0 < K; k0 += GB_KT) {
        int hd  = k0 >> 12;         // / 4096
        int rem = k0 & 4095;
        int ii  = rem >> 6;         // / 64
        int j0  = rem & 63;         // 0 or 32
        // ---- W tile: 64 cols x 8 k-chunks -> 4 chunks/thread ----
        #pragma unroll
        for (int it = 0; it < 4; ++it) {
            int i = tid + it * 128;
            int c = i >> 3, kc = (i & 7) * 4;
            int gc = col_base + c;
            float v0 = 0.f, v1 = 0.f, v2 = 0.f, v3 = 0.f;
            if (gc < R) {
                const float* wp = W + (long long)(k0 + kc) * R + gc;
                v0 = wp[0]; v1 = wp[R]; v2 = wp[2 * R]; v3 = wp[3 * R];
                if (k0 + GB_KT < K) __builtin_prefetch(wp + (long long)GB_KT * R, 0, 3);
            }
            h4 h = { (_Float16)v0, (_Float16)v1, (_Float16)v2, (_Float16)v3 };
            *(h4*)&Wp[c][perm_idx(kc)] = h;
        }
        // ---- X tile: 16 rows x 8 k-chunks = 128 chunks -> 1/thread ----
        {
            int r = tid >> 3, kc = (tid & 7) * 4;
            _Float16 a1v = A1s[r][hd * 64 + ii];
            h4 a2v = *(const h4*)&A2s[r][hd * 64 + j0 + kc];
            h4 x = { a1v * a2v[0], a1v * a2v[1], a1v * a2v[2], a1v * a2v[3] };
            *(h4*)&Xp[r][perm_idx(kc)] = x;
        }
        __syncthreads();
        v16h af = frag_load(&Xp[l15][grp * 16]);
        v16h bf = frag_load(&Wp[wave * 16 + l15][grp * 16]);
        acc = __builtin_amdgcn_wmma_f32_16x16x32_f16(false, af, false, bf,
                                                     (short)0, acc, false, false);
        __syncthreads();
    }
    #pragma unroll
    for (int r = 0; r < 8; ++r) {
        int gm = row_base + grp * 8 + r;
        int gn = col_base + wave * 16 + l15;
        if (gm < NrowsArg && gn < R)
            out[(long long)gm * R + gn] = acc[r] + bias[gn];
    }
}

// ---------------- glue kernels ----------------
__global__ __launch_bounds__(256)
void k_concat(const float* __restrict__ nodes, const int* __restrict__ hts, int sel,
              const float* __restrict__ right, float* __restrict__ out) {
    int b = blockIdx.x / Pn, p = blockIdx.x % Pn;
    int e = hts[sel * Pn + p];
    const float* ns = nodes + ((long long)b * NEn + e) * Dn;
    const float* rr = right + (long long)(b * Pn + p) * Dn;
    float* o = out + (long long)(b * Pn + p) * (2 * Dn);
    for (int d = threadIdx.x; d < Dn; d += blockDim.x) {
        o[d] = ns[d];
        o[Dn + d] = rr[d];
    }
}

__global__ void k_mask(const float* __restrict__ bin, int* __restrict__ mask, int n) {
    int i = blockIdx.x * blockDim.x + threadIdx.x;
    if (i < n) mask[i] = (bin[i * 2 + 1] > bin[i * 2 + 0]) ? 1 : 0;
}

__global__ __launch_bounds__(256)
void k_fout(const float* __restrict__ niP, const float* __restrict__ njP,
            const float* __restrict__ efW, const float* __restrict__ egat_b,
            const int* __restrict__ hts, float* __restrict__ fout) {
    int b = blockIdx.x / En, e = blockIdx.x % En;
    int s  = (e < Pn) ? hts[e] : (e - Pn);
    int dd = (e < Pn) ? hts[Pn + e] : (e - Pn);
    const float* fi = niP + ((long long)b * NEn + s) * Dn;
    const float* fj = njP + ((long long)b * NEn + dd) * Dn;
    const float* fe = (e < Pn) ? (efW + (long long)(b * Pn + e) * Dn) : nullptr;
    float* o = fout + (long long)(b * En + e) * Dn;
    for (int d = threadIdx.x; d < Dn; d += blockDim.x)
        o[d] = fi[d] + fj[d] + (fe ? fe[d] : 0.f) + egat_b[d];
}

__global__ __launch_bounds__(32)
void k_score(const float* __restrict__ fout, const float* __restrict__ attn_p,
             const int* __restrict__ mask, float* __restrict__ score) {
    int idx = blockIdx.x;
    int hh = idx % Hn, e = (idx / Hn) % En, b = idx / (Hn * En);
    int lane = threadIdx.x;
    const float* f = fout + (long long)(b * En + e) * Dn + hh * BSn;
    float s = 0.f;
    for (int k = lane; k < BSn; k += 32) {
        float v = f[k];
        v = (v > 0.f) ? v : 0.2f * v;        // leaky_relu, NEG=0.2
        s += v * attn_p[hh * BSn + k];
    }
    for (int off = 16; off > 0; off >>= 1) s += __shfl_xor(s, off, 32);
    if (lane == 0) {
        int m = (e < Pn) ? mask[b * Pn + e] : 1;
        score[(b * En + e) * Hn + hh] = m ? s : -1e30f;
    }
}

__global__ __launch_bounds__(32)
void k_seg(const float* __restrict__ score, const int* __restrict__ hts,
           const int* __restrict__ mask, float* __restrict__ smax,
           float* __restrict__ den) {
    int b = blockIdx.x / NEn, n = blockIdx.x % NEn;
    int hh = threadIdx.x;
    if (hh >= Hn) return;
    float mx = -3e38f;
    for (int e = 0; e < En; ++e) {
        int dd = (e < Pn) ? hts[Pn + e] : (e - Pn);
        if (dd == n) mx = fmaxf(mx, score[(b * En + e) * Hn + hh]);
    }
    float dn = 0.f;
    for (int e = 0; e < En; ++e) {
        int dd = (e < Pn) ? hts[Pn + e] : (e - Pn);
        if (dd == n) {
            int m = (e < Pn) ? mask[b * Pn + e] : 1;
            if (m) dn += __expf(score[(b * En + e) * Hn + hh] - mx);
        }
    }
    smax[(b * NEn + n) * Hn + hh] = mx;
    den[(b * NEn + n) * Hn + hh] = dn;
}

__global__ void k_alpha(const float* __restrict__ score, const float* __restrict__ smax,
                        const float* __restrict__ den, const int* __restrict__ hts,
                        const int* __restrict__ mask, float* __restrict__ alpha) {
    int i = blockIdx.x * blockDim.x + threadIdx.x;
    if (i >= Bn * En * Hn) return;
    int hh = i % Hn, e = (i / Hn) % En, b = i / (Hn * En);
    int dd = (e < Pn) ? hts[Pn + e] : (e - Pn);
    int m  = (e < Pn) ? mask[b * Pn + e] : 1;
    float ex = m ? __expf(score[i] - smax[(b * NEn + dd) * Hn + hh]) : 0.f;
    alpha[i] = ex / (den[(b * NEn + dd) * Hn + hh] + 1e-20f);
}

__global__ __launch_bounds__(256)
void k_aggr(const float* __restrict__ nodeP, const float* __restrict__ alpha,
            const int* __restrict__ hts, float* __restrict__ nodeout) {
    int b = blockIdx.x / NEn, n = blockIdx.x % NEn;
    for (int d = threadIdx.x; d < Dn; d += blockDim.x) {
        int hh = d / BSn;
        float acc = 0.f;
        for (int e = 0; e < En; ++e) {
            int dd = (e < Pn) ? hts[Pn + e] : (e - Pn);
            if (dd != n) continue;
            int s = (e < Pn) ? hts[e] : (e - Pn);
            acc += nodeP[((long long)b * NEn + s) * Dn + d] * alpha[(b * En + e) * Hn + hh];
        }
        nodeout[((long long)b * NEn + n) * Dn + d] = acc;
    }
}

__global__ void k_newci(const float* __restrict__ fout, const float* __restrict__ ci,
                        const int* __restrict__ mask, float* __restrict__ nci) {
    long long i = (long long)blockIdx.x * blockDim.x + threadIdx.x;
    if (i >= (long long)Bn * Pn * Dn) return;
    int d = (int)(i % Dn);
    long long bp = i / Dn;
    int b = (int)(bp / Pn), p = (int)(bp % Pn);
    nci[i] = mask[bp] ? fout[((long long)b * En + p) * Dn + d] : ci[i];
}

// ---------------- host launch ----------------
extern "C" void kernel_launch(void* const* d_in, const int* in_sizes, int n_in,
                              void* d_out, int out_size, void* d_ws, size_t ws_size,
                              hipStream_t stream) {
    (void)in_sizes; (void)n_in; (void)out_size; (void)ws_size;
    const float* context = (const float*)d_in[0];
    const float* att     = (const float*)d_in[1];
    const int*   mmap    = (const int*)  d_in[2];
    const float* emap    = (const float*)d_in[3];
    const int*   hts     = (const int*)  d_in[4];
    const float* Whb  = (const float*)d_in[5];  const float* bhb  = (const float*)d_in[6];
    const float* Wtb  = (const float*)d_in[7];  const float* btb  = (const float*)d_in[8];
    const float* Wbin = (const float*)d_in[9];  const float* bbin = (const float*)d_in[10];
    const float* Wrel = (const float*)d_in[11]; const float* brel = (const float*)d_in[12];
    const float* Wh   = (const float*)d_in[13]; const float* bh   = (const float*)d_in[14];
    const float* Wt   = (const float*)d_in[15]; const float* bt   = (const float*)d_in[16];
    const float* Wnode= (const float*)d_in[17]; const float* Wni  = (const float*)d_in[18];
    const float* Wfij = (const float*)d_in[19]; const float* Wnj  = (const float*)d_in[20];
    const float* attn_p = (const float*)d_in[21];
    const float* egat_b = (const float*)d_in[22];
    float* out = (float*)d_out;

    float* wsf = (float*)d_ws;
    size_t off = 0;
    auto alloc = [&](size_t n) { float* p = wsf + off; off += n; return p; };
    float* entity  = alloc((size_t)Bn * NEn * Dn);
    float* esum    = alloc(128);
    float* cur     = alloc((size_t)Bn * Hn * NEn * Ln);
    float* ca      = alloc((size_t)Bn * Pn * Ln);          // reused later as new_ci
    float* ci      = alloc((size_t)Nrow * Dn);
    float* catA    = alloc((size_t)Nrow * 2 * Dn);
    float* act1    = alloc((size_t)Nrow * Dn);
    float* act2    = alloc((size_t)Nrow * Dn);
    int*   maskb   = (int*)alloc(2304);
    float* niP     = alloc((size_t)Bn * NEn * Dn);
    float* njP     = alloc((size_t)Bn * NEn * Dn);
    float* nodeP   = alloc((size_t)Bn * NEn * Dn);
    float* efW     = alloc((size_t)Nrow * Dn);
    float* fout    = alloc((size_t)Bn * En * Dn);
    float* score   = alloc((size_t)Bn * En * Hn);
    float* smax    = alloc((size_t)Bn * NEn * Hn);
    float* den     = alloc((size_t)Bn * NEn * Hn);
    float* alpha   = alloc((size_t)Bn * En * Hn);
    float* nodeout = alloc((size_t)Bn * NEn * Dn);
    float* new_ci  = ca;   // ca dead after ci GEMM; region is large enough

    // 1) entity pooling
    k_entity<<<Bn * NEn, 256, 0, stream>>>(context, mmap, emap, entity, esum);
    // 2) cur
    k_cur<<<Bn * Hn * NEn, 256, 0, stream>>>(att, mmap, emap, esum, cur);
    // 3) ca
    k_ca<<<Bn * Pn, 256, 0, stream>>>(cur, hts, ca);
    // 4) ci = ca @ context   (batched WMMA GEMM)
    k_gemm<<<dim3(Dn / 64, (Pn + 63) / 64, Bn), 256, 0, stream>>>(
        ca, context, nullptr, ci, Pn, Ln, Dn,
        (long long)Pn * Ln, (long long)Ln * Dn, (long long)Pn * Dn, 0);
    // 5) bin_h = tanh([h|ci] @ Whb + bhb)
    k_concat<<<Bn * Pn, 256, 0, stream>>>(entity, hts, 0, ci, catA);
    k_gemm<<<dim3(Dn / 64, (Nrow + 63) / 64, 1), 256, 0, stream>>>(
        catA, Whb, bhb, act1, Nrow, 2 * Dn, Dn, 0, 0, 0, 1);
    // 6) bin_t
    k_concat<<<Bn * Pn, 256, 0, stream>>>(entity, hts, 1, ci, catA);
    k_gemm<<<dim3(Dn / 64, (Nrow + 63) / 64, 1), 256, 0, stream>>>(
        catA, Wtb, btb, act2, Nrow, 2 * Dn, Dn, 0, 0, 0, 1);
    // 7) bin_res -> d_out[0 : 2208*2]  (fused bilinear WMMA)
    k_gbl<<<dim3(1, (Nrow + 15) / 16), 128, 0, stream>>>(
        act1, act2, Wbin, bbin, out, Nrow, 2);
    // 8) mask = argmax(bin)==1
    k_mask<<<(Nrow + 255) / 256, 256, 0, stream>>>(out, maskb, Nrow);
    // 9) node / edge projections (WMMA)
    k_gemm<<<dim3(Dn / 64, 2, 1), 256, 0, stream>>>(
        entity, Wni, nullptr, niP, Bn * NEn, Dn, Dn, 0, 0, 0, 0);
    k_gemm<<<dim3(Dn / 64, 2, 1), 256, 0, stream>>>(
        entity, Wnj, nullptr, njP, Bn * NEn, Dn, Dn, 0, 0, 0, 0);
    k_gemm<<<dim3(Dn / 64, 2, 1), 256, 0, stream>>>(
        entity, Wnode, nullptr, nodeP, Bn * NEn, Dn, Dn, 0, 0, 0, 0);
    k_gemm<<<dim3(Dn / 64, (Nrow + 63) / 64, 1), 256, 0, stream>>>(
        ci, Wfij, nullptr, efW, Nrow, Dn, Dn, 0, 0, 0, 0);
    // 10) EGAT edge features + attention
    k_fout<<<Bn * En, 256, 0, stream>>>(niP, njP, efW, egat_b, hts, fout);
    k_score<<<Bn * En * Hn, 32, 0, stream>>>(fout, attn_p, maskb, score);
    k_seg<<<Bn * NEn, 32, 0, stream>>>(score, hts, maskb, smax, den);
    k_alpha<<<(Bn * En * Hn + 255) / 256, 256, 0, stream>>>(
        score, smax, den, hts, maskb, alpha);
    k_aggr<<<Bn * NEn, 256, 0, stream>>>(nodeP, alpha, hts, nodeout);
    // 11) new_ci = mask ? ef_new : ci
    k_newci<<<(int)(((long long)Bn * Pn * Dn + 255) / 256), 256, 0, stream>>>(
        fout, ci, maskb, new_ci);
    // 12) nh / nt
    k_concat<<<Bn * Pn, 256, 0, stream>>>(nodeout, hts, 0, new_ci, catA);
    k_gemm<<<dim3(Dn / 64, (Nrow + 63) / 64, 1), 256, 0, stream>>>(
        catA, Wh, bh, act1, Nrow, 2 * Dn, Dn, 0, 0, 0, 1);
    k_concat<<<Bn * Pn, 256, 0, stream>>>(nodeout, hts, 1, new_ci, catA);
    k_gemm<<<dim3(Dn / 64, (Nrow + 63) / 64, 1), 256, 0, stream>>>(
        catA, Wt, bt, act2, Nrow, 2 * Dn, Dn, 0, 0, 0, 1);
    // 13) relation_res -> d_out[2208*2 : ]
    k_gbl<<<dim3((Rn + 63) / 64, (Nrow + 15) / 16), 128, 0, stream>>>(
        act1, act2, Wrel, brel, out + (size_t)Nrow * 2, Nrow, Rn);
}
